// Mamba2_84653805404786
// MI455X (gfx1250) — matrix operand order repaired and checked by reading, not compiled, not run on hardware
//
#include <hip/hip_runtime.h>
#include <math.h>

// ---------------- problem constants ----------------
#define D_MODEL   1024
#define D_STATE   128
#define D_CONV    4
#define HEADDIM   64
#define CHUNK     256
#define BATCH     2
#define SEQLEN    2048
#define D_INNER   2048
#define D_SSM     2048
#define NHEADS    32
#define CONV_DIM  2304          // D_SSM + 2*D_STATE
#define D_IN_PROJ 4384          // 2*D_INNER + 2*D_STATE + NHEADS
#define NPROJ_PAD 4480          // D_IN_PROJ rounded up to 128 (zero-padded weights)
#define NC        8             // SEQLEN / CHUNK
#define BL        (BATCH * SEQLEN)   // 4096 tokens

// ---------------- WMMA plumbing ----------------
typedef __attribute__((ext_vector_type(16))) _Float16 v16h;
typedef __attribute__((ext_vector_type(8)))  _Float16 v8h;
typedef __attribute__((ext_vector_type(8)))  float    v8f;
typedef __attribute__((ext_vector_type(4)))  unsigned int v4u;
typedef __attribute__((ext_vector_type(8)))  int v8i;
typedef __attribute__((ext_vector_type(4)))  int v4i;

__device__ __forceinline__ v8f wmma32(v16h a, v16h b, v8f c) {
  return __builtin_amdgcn_wmma_f32_16x16x32_f16(
      false, a, false, b, (short)0, c, false, false);
}

// A fragment (16x32 f16), ISA interleaved layout:
// lane<16 : row=lane,    K in {0..7, 16..23};  lane>=16: row=lane-16, K in {8..15, 24..31}
__device__ __forceinline__ v16h load_a_frag(const _Float16* base, int stride) {
  const int lane = threadIdx.x & 31;
  const int lr = lane & 15, hi = lane >> 4;
  const _Float16* p = base + lr * stride + hi * 8;
  v8h lo = *(const v8h*)p;
  v8h hh = *(const v8h*)(p + 16);
  v16h r;
#pragma unroll
  for (int i = 0; i < 8; ++i) { r[i] = lo[i]; r[i + 8] = hh[i]; }
  return r;
}

// B fragment (32x16 f16) from B^T stored [n][K]: lane<16: col=lane, K in {0..7, 16..23}; lane>=16: K in {8..15, 24..31}
__device__ __forceinline__ v16h load_b_frag(const _Float16* base, int stride) {
  const int lane = threadIdx.x & 31;
  const int lr = lane & 15, hi = lane >> 4;
  const _Float16* p = base + lr * stride + hi * 8;
  v8h lo = *(const v8h*)p;
  v8h hh = *(const v8h*)(p + 16);
  v16h r;
#pragma unroll
  for (int i = 0; i < 8; ++i) { r[i] = lo[i]; r[i + 8] = hh[i]; }
  return r;
}

// ---------------- CDNA5 async global->LDS DMA (ASYNCcnt) ----------------
__device__ __forceinline__ void async_b128(void* lds_dst, const void* gsrc) {
  uint32_t lds_off = (uint32_t)(uintptr_t)lds_dst;   // low 32 bits of generic = LDS offset
  asm volatile("global_load_async_to_lds_b128 %0, %1, off"
               :: "v"(lds_off), "v"(gsrc) : "memory");
}
__device__ __forceinline__ void async_wait0() {
  asm volatile("s_wait_asynccnt 0x0" ::: "memory");
}

// ---------------- CDNA5 Tensor Data Mover (TENSORcnt) ----------------
// 2D tile load: tile_dim0=32 f16 (64B rows), tile_dim1=128 rows, stride=K
// elements. LDS padding done by TDM: pad_interval=3 (every 16 DWORDs = one
// row) pad_amount=3 (4 DWORDs = 16B) -> 80B LDS row stride == LK halves.
#define LK 40
__device__ __forceinline__ void tdm_load_tile(void* lds_dst, const _Float16* gsrc, int K) {
  const uint64_t ga = (uint64_t)(uintptr_t)gsrc;
  const uint32_t la = (uint32_t)(uintptr_t)lds_dst;
  v4u g0;
  g0[0] = 1u;                                              // count=1, user descriptor
  g0[1] = la;                                              // lds_addr [63:32]
  g0[2] = (uint32_t)ga;                                    // global_addr low
  g0[3] = (uint32_t)((ga >> 32) & 0x01FFFFFFu) | (2u << 30);  // addr[56:32] | type=2
  v8i g1;
  g1[0] = (int)((1u << 16) | (1u << 20) | (3u << 22) | (3u << 25));
        // data_size=2B | pad_enable | pad_interval=16DW | pad_amount=4DW
  g1[1] = (int)(((uint32_t)K & 0xFFFFu) << 16);            // tensor_dim0[15:0] @ [63:48]
  g1[2] = (int)(((uint32_t)K >> 16) | (128u << 16));       // tensor_dim0 hi | tensor_dim1=128
  g1[3] = (int)(32u << 16);                                // tensor_dim1 hi=0 | tile_dim0=32
  g1[4] = (int)128u;                                       // tile_dim1=128 | tile_dim2=0
  g1[5] = (int)K;                                          // tensor_dim0_stride[31:0]
  g1[6] = 0;                                               // stride hi | dim1_stride lo
  g1[7] = 0;
  const v4i gz4 = {0, 0, 0, 0};
  const v8i gz8 = {0, 0, 0, 0, 0, 0, 0, 0};
  __builtin_amdgcn_tensor_load_to_lds(g0, g1, gz4, gz4, gz8, 0);
}

// =====================================================================
// K0: f32 -> f16 conversion with optional zero tail padding
// =====================================================================
__global__ __launch_bounds__(256) void k_cvt(
    const float* __restrict__ src, _Float16* __restrict__ dst,
    long nsrc, long ndst)
{
  const long i = (long)blockIdx.x * 256 + threadIdx.x;
  if (i < ndst) dst[i] = (i < nsrc) ? (_Float16)src[i] : (_Float16)0.f;
}

// =====================================================================
// K1/K10: GEMM  C[M,N] = A[M,K] @ W[N,K]^T, f16 operands, TDM-staged
// double-buffered LDS tiles, 128x128x32, 8 waves, 8 WMMA frags/wave.
// N grid dimension must cover multiples of 128 (weights zero-padded);
// stores guarded by Nstore.
// =====================================================================
__global__ __launch_bounds__(256) void k_gemm16(
    const _Float16* __restrict__ A, const _Float16* __restrict__ W,
    float* __restrict__ C, int M, int Nstore, int K)
{
  __shared__ __align__(16) _Float16 As[2][128 * LK];
  __shared__ __align__(16) _Float16 Ws[2][128 * LK];
  const int tid = threadIdx.x, lane = tid & 31, wave = tid >> 5;
  const int bm = blockIdx.y * 128, bn = blockIdx.x * 128;
  const int wm = (wave & 1) * 64, wn = (wave >> 1) * 32;

  v8f acc[4][2];
#pragma unroll
  for (int i = 0; i < 4; ++i)
#pragma unroll
    for (int j = 0; j < 2; ++j) acc[i][j] = {};

  const int NT = K / 32;

  // wave 0 moves the A tile, wave 1 the W tile (TDM: one issue per tile)
  auto issue = [&](int kt, int buf) {
    if (wave == 0) tdm_load_tile(&As[buf][0], A + (size_t)bm * K + kt * 32, K);
    else if (wave == 1) tdm_load_tile(&Ws[buf][0], W + (size_t)bn * K + kt * 32, K);
  };

  auto compute = [&](int cur) {
    v16h af[4], wf[2];
#pragma unroll
    for (int i = 0; i < 4; ++i) af[i] = load_a_frag(&As[cur][(wm + i * 16) * LK], LK);
#pragma unroll
    for (int j = 0; j < 2; ++j) wf[j] = load_b_frag(&Ws[cur][(wn + j * 16) * LK], LK);
#pragma unroll
    for (int i = 0; i < 4; ++i)
#pragma unroll
      for (int j = 0; j < 2; ++j) acc[i][j] = wmma32(af[i], wf[j], acc[i][j]);
  };

  issue(0, 0);
  for (int it = 0; it < NT - 1; ++it) {
    const int cur = it & 1;
    issue(it + 1, cur ^ 1);                       // prefetch next tile
    __builtin_amdgcn_s_wait_tensorcnt(1);         // tile `it` landed (per-wave)
    __syncthreads();
    compute(cur);
    __syncthreads();                              // reads done before re-target
  }
  __builtin_amdgcn_s_wait_tensorcnt(0);
  __syncthreads();
  compute((NT - 1) & 1);

  const int lr = lane & 15, hi = lane >> 4;
#pragma unroll
  for (int i = 0; i < 4; ++i)
#pragma unroll
    for (int j = 0; j < 2; ++j) {
      const int col = bn + wn + j * 16 + lr;
      if (col < Nstore) {
#pragma unroll
        for (int r = 0; r < 8; ++r) {
          const int row = bm + wm + i * 16 + hi * 8 + r;
          C[(size_t)row * Nstore + col] = acc[i][j][r];
        }
      }
    }
}

// =====================================================================
// K2: depthwise causal conv (K=4) + SiLU; emits f16 x and f16 B|C
// =====================================================================
__global__ __launch_bounds__(256) void k_conv(
    const float* __restrict__ zxbcdt, const float* __restrict__ conv_w,
    const float* __restrict__ conv_b, _Float16* __restrict__ x16,
    _Float16* __restrict__ bc16)
{
  const int idx = blockIdx.x * 256 + threadIdx.x;     // over BL*CONV_DIM
  const int c = idx % CONV_DIM;
  const int t = idx / CONV_DIM;
  const int l = t % SEQLEN, b = t / SEQLEN;
  float acc = conv_b[c];
#pragma unroll
  for (int j = 0; j < D_CONV; ++j) {
    const int lp = l - (D_CONV - 1) + j;
    if (lp >= 0)
      acc += conv_w[c * D_CONV + j] *
             zxbcdt[((size_t)(b * SEQLEN + lp)) * D_IN_PROJ + D_SSM + c];
  }
  const float v = acc / (1.f + __expf(-acc));
  if (c < D_SSM) x16[(size_t)t * D_SSM + c] = (_Float16)v;
  else           bc16[(size_t)t * 256 + (c - D_SSM)] = (_Float16)v;  // B: 0..127, C: 128..255
}

// =====================================================================
// K3: dt softplus + per-chunk cumsum of dA (Hillis-Steele in LDS)
// =====================================================================
__global__ __launch_bounds__(256) void k_dt_scan(
    const float* __restrict__ zxbcdt, const float* __restrict__ dt_bias,
    const float* __restrict__ A_log, float* __restrict__ dtws,
    float* __restrict__ dacs, float* __restrict__ csum)
{
  int bid = blockIdx.x;
  const int c = bid & 7; bid >>= 3;
  const int h = bid & 31; const int b = bid >> 5;
  const int t = threadIdx.x;
  const size_t tok = (size_t)b * SEQLEN + c * CHUNK + t;
  float raw = zxbcdt[tok * D_IN_PROJ + (D_SSM + CONV_DIM) + h] + dt_bias[h];
  float dtv = (raw > 20.f) ? raw : log1pf(__expf(raw));
  dtws[tok * NHEADS + h] = dtv;
  float da = dtv * (-__expf(A_log[h]));

  __shared__ float sc[CHUNK];
  sc[t] = da;
  __syncthreads();
  for (int off = 1; off < CHUNK; off <<= 1) {
    float v = (t >= off) ? sc[t - off] : 0.f;
    __syncthreads();
    sc[t] += v;
    __syncthreads();
  }
  const size_t ob = ((size_t)(b * NHEADS + h) * NC + c) * CHUNK;
  dacs[ob + t] = sc[t];
  if (t == CHUNK - 1) csum[(b * NHEADS + h) * NC + c] = sc[t];
}

// =====================================================================
// K4: Xdt^T[(b,c,h)][p][l] = x[b,l,h,p]*dt[b,l,h]  (f16, K-contiguous)
// =====================================================================
__global__ __launch_bounds__(256) void k_xdtT(
    const _Float16* __restrict__ x16, const float* __restrict__ dtws,
    _Float16* __restrict__ xdt16)
{
  size_t idx = (size_t)blockIdx.x * 256 + threadIdx.x;
  const int l = idx & 255; size_t r = idx >> 8;
  const int p = r & 63; r >>= 6;
  const int h = r & 31; r >>= 5;
  const int c = r & 7;  const int b = (int)(r >> 3);
  const size_t tok = (size_t)b * SEQLEN + c * CHUNK + l;
  xdt16[idx] = (_Float16)((float)x16[tok * D_SSM + h * HEADDIM + p] *
                          dtws[tok * NHEADS + h]);
}

// =====================================================================
// K5: local chunk states  local[p,n] = sum_l (Xdt[l,p]*decay[l]) B[l,n]
// =====================================================================
__global__ __launch_bounds__(256) void k_chunk_state(
    const _Float16* __restrict__ bc16, const _Float16* __restrict__ xdt16,
    const float* __restrict__ dacs, const float* __restrict__ csum,
    float* __restrict__ localS)
{
  int bid = blockIdx.x;                      // ((b*NC+c)*NHEADS+h)
  const int h = bid & 31; const int c = (bid >> 5) & 7; const int b = bid >> 8;
  __shared__ __align__(16) _Float16 As[64 * LK];    // [p][l-step], decay-scaled
  __shared__ __align__(16) _Float16 Ws[128 * LK];   // [n][l-step] (B^T)
  __shared__ float dA[CHUNK];
  const int tid = threadIdx.x, lane = tid & 31, wave = tid >> 5;

  dA[tid] = dacs[((size_t)(b * NHEADS + h) * NC + c) * CHUNK + tid];
  const float cs = csum[(b * NHEADS + h) * NC + c];
  __syncthreads();

  const size_t tokbase = (size_t)b * SEQLEN + c * CHUNK;
  const _Float16* xd = xdt16 + (size_t)blockIdx.x * 64 * 256;
  const int wp = wave & 1, wn = wave >> 1;

  v8f acc[2][2];
#pragma unroll
  for (int i = 0; i < 2; ++i)
#pragma unroll
    for (int j = 0; j < 2; ++j) acc[i][j] = {};

  for (int kk = 0; kk < 8; ++kk) {
    {   // stage A (needs decay multiply -> VALU path)
      const int p = tid >> 2, j0 = (tid & 3) * 8;
#pragma unroll
      for (int j = 0; j < 8; ++j) {
        const int l = kk * 32 + j0 + j;
        As[p * LK + j0 + j] =
            (_Float16)((float)xd[(size_t)p * 256 + l] * __expf(cs - dA[l]));
      }
    }
    {   // stage W = B^T (transposed scatter of f16 B rows)
      const int lloc = tid >> 3, n0 = (tid & 7) * 16;
      const _Float16* src = bc16 + (tokbase + kk * 32 + lloc) * 256 + n0;
#pragma unroll
      for (int j = 0; j < 16; ++j) Ws[(n0 + j) * LK + lloc] = src[j];
    }
    __syncthreads();
    v16h af[2], wf[2];
#pragma unroll
    for (int i = 0; i < 2; ++i) af[i] = load_a_frag(&As[(wp * 32 + i * 16) * LK], LK);
#pragma unroll
    for (int j = 0; j < 2; ++j) wf[j] = load_b_frag(&Ws[(wn * 32 + j * 16) * LK], LK);
#pragma unroll
    for (int i = 0; i < 2; ++i)
#pragma unroll
      for (int j = 0; j < 2; ++j) acc[i][j] = wmma32(af[i], wf[j], acc[i][j]);
    __syncthreads();
  }

  float* dst = localS + (size_t)blockIdx.x * 64 * 128;
  const int lr = lane & 15, hi = lane >> 4;
#pragma unroll
  for (int i = 0; i < 2; ++i)
#pragma unroll
    for (int j = 0; j < 2; ++j) {
      const int n = wn * 32 + j * 16 + lr;
#pragma unroll
      for (int r = 0; r < 8; ++r) {
        const int p = wp * 32 + i * 16 + hi * 8 + r;
        dst[(size_t)p * 128 + n] = acc[i][j][r];
      }
    }
}

// =====================================================================
// K6: inter-chunk recurrence  S_{c+1} = exp(csum_c)*S_c + local_c
// =====================================================================
__global__ __launch_bounds__(256) void k_state_scan(
    const float* __restrict__ localS, const float* __restrict__ csum,
    _Float16* __restrict__ prev16)
{
  const int b = blockIdx.x >> 5, h = blockIdx.x & 31;
  const int t = threadIdx.x;
  float s[32];
#pragma unroll
  for (int k = 0; k < 32; ++k) s[k] = 0.f;
  for (int c = 0; c < NC; ++c) {
    const size_t base = ((size_t)((b * NC + c) * NHEADS + h)) * 64 * 128;
#pragma unroll
    for (int k = 0; k < 32; ++k) prev16[base + t + k * 256] = (_Float16)s[k];
    const float dec = __expf(csum[(b * NHEADS + h) * NC + c]);
#pragma unroll
    for (int k = 0; k < 32; ++k) s[k] = s[k] * dec + localS[base + t + k * 256];
  }
}

// =====================================================================
// K7: per-(b,c,h, l-tile of 64):
//   G = C·B^T (WMMA) -> S[l,s]=G*exp(dA[l]-dA[s]) masked -> f16 LDS
//   Y = S·Xdt + exp(dA[l])*(C·prev^T) + D*x   (WMMA; async-DMA staging)
// =====================================================================
__global__ __launch_bounds__(256) void k_chunk_y(
    const _Float16* __restrict__ bc16, const _Float16* __restrict__ xdt16,
    const _Float16* __restrict__ prev16, const _Float16* __restrict__ x16,
    const float* __restrict__ dacs, const float* __restrict__ Dp,
    float* __restrict__ yws)
{
  int bid = blockIdx.x;
  const int q = bid & 3;  bid >>= 2;          // l-tile (0..3)
  const int h = bid & 31; bid >>= 5;
  const int c = bid & 7;  const int b = bid >> 3;
  const int lt0 = q * 64;

  __shared__ __align__(16) _Float16 Cs[64 * 136];   // [l][n]
  __shared__ __align__(16) _Float16 Ss[64 * 264];   // [l][s] masked scores
  __shared__ __align__(16) _Float16 Wb[32 * 136];   // staging (B rows / W tiles)
  __shared__ float dA[CHUNK];

  const int tid = threadIdx.x, lane = tid & 31, wave = tid >> 5;
  const size_t tokbase = (size_t)b * SEQLEN + c * CHUNK;
  const size_t bch = (size_t)((b * NC + c) * NHEADS + h);

  dA[tid] = dacs[((size_t)(b * NHEADS + h) * NC + c) * CHUNK + tid];
  {   // stage Cs (64 rows l, 128 halves) via async DMA; C lives at +128
    const int r = tid >> 2;
    const int s4 = (tid & 3) * 4;
    const _Float16* src = bc16 + (tokbase + lt0 + r) * 256 + 128;
#pragma unroll
    for (int i = 0; i < 4; ++i)
      async_b128(&Cs[r * 136 + (s4 + i) * 8], src + (s4 + i) * 8);
  }
  async_wait0();
  __syncthreads();

  const int lt = wave & 3;            // l sub-tile (x16)
  const int sh = wave >> 2;           // s sub-tile within 32-stage
  const int lr = lane & 15, hi = lane >> 4;

  // ---- phase G: masked scores into Ss ----
  for (int st = 0; st < 8; ++st) {
    {   // stage B rows [32 s][128 halves] via async DMA
      const int r = tid >> 3;
      const int sg = (tid & 7) * 2;
      const _Float16* src = bc16 + (tokbase + st * 32 + r) * 256;
#pragma unroll
      for (int i = 0; i < 2; ++i)
        async_b128(&Wb[r * 136 + (sg + i) * 8], src + (sg + i) * 8);
    }
    async_wait0();
    __syncthreads();
    v8f g = {};
#pragma unroll
    for (int ks = 0; ks < 4; ++ks) {
      v16h a = load_a_frag(&Cs[(lt * 16) * 136 + ks * 32], 136);
      v16h w = load_b_frag(&Wb[(sh * 16) * 136 + ks * 32], 136);
      g = wmma32(a, w, g);
    }
    const int s_loc = st * 32 + sh * 16 + lr;
#pragma unroll
    for (int r8 = 0; r8 < 8; ++r8) {
      const int lrow = lt * 16 + hi * 8 + r8;
      const int l_loc = lt0 + lrow;
      float val = 0.f;
      if (l_loc >= s_loc) val = g[r8] * __expf(dA[l_loc] - dA[s_loc]);
      Ss[lrow * 264 + s_loc] = (_Float16)val;
    }
    __syncthreads();
  }

  // ---- phases B & C: Y_off (C·prev^T) and Y_diag (S·Xdt) ----
  const int pq = wave >> 2;           // p half (0/1)
  v8f yoff[2] = {{}, {}}, ydia[2] = {{}, {}};
  const int sp = tid >> 2, j0 = (tid & 3) * 8;

  for (int kn = 0; kn < 4; ++kn) {    // Y_off over K = n (128)
    async_b128(&Wb[sp * LK + j0],
               prev16 + bch * 64 * 128 + (size_t)sp * 128 + kn * 32 + j0);
    async_wait0();
    __syncthreads();
    v16h a = load_a_frag(&Cs[(lt * 16) * 136 + kn * 32], 136);
#pragma unroll
    for (int f = 0; f < 2; ++f) {
      v16h w = load_b_frag(&Wb[(pq * 32 + f * 16) * LK], LK);
      yoff[f] = wmma32(a, w, yoff[f]);
    }
    __syncthreads();
  }
  const _Float16* xd = xdt16 + bch * 64 * 256;
  for (int ks = 0; ks < 8; ++ks) {    // Y_diag over K = s (256)
    async_b128(&Wb[sp * LK + j0], xd + (size_t)sp * 256 + ks * 32 + j0);
    async_wait0();
    __syncthreads();
    v16h a = load_a_frag(&Ss[(lt * 16) * 264 + ks * 32], 264);
#pragma unroll
    for (int f = 0; f < 2; ++f) {
      v16h w = load_b_frag(&Wb[(pq * 32 + f * 16) * LK], LK);
      ydia[f] = wmma32(a, w, ydia[f]);
    }
    __syncthreads();
  }

  // ---- epilogue: y = Y_diag + exp(dA[l])*Y_off + D*x ----
  const float Dv = Dp[h];
#pragma unroll
  for (int f = 0; f < 2; ++f) {
    const int p = pq * 32 + f * 16 + lr;
#pragma unroll
    for (int r8 = 0; r8 < 8; ++r8) {
      const int l_loc = lt0 + lt * 16 + hi * 8 + r8;
      const size_t tok = tokbase + l_loc;
      const float xv = (float)x16[tok * D_SSM + h * HEADDIM + p];
      yws[tok * D_SSM + h * HEADDIM + p] =
          ydia[f][r8] + __expf(dA[l_loc]) * yoff[f][r8] + Dv * xv;
    }
  }
}

// =====================================================================
// K8: gating y*silu(z) + RMSNorm -> f16 activations for out_proj GEMM
// =====================================================================
__global__ __launch_bounds__(256) void k_gate_norm(
    const float* __restrict__ yws, const float* __restrict__ zxbcdt,
    const float* __restrict__ norm_w, _Float16* __restrict__ yg16)
{
  const size_t g = blockIdx.x;
  const int t = threadIdx.x;
  float v[8]; float ss = 0.f;
#pragma unroll
  for (int k = 0; k < 8; ++k) {
    const int ch = t + k * 256;
    const float y = yws[g * D_SSM + ch];
    const float z = zxbcdt[g * D_IN_PROJ + ch];
    const float yg = y * (z / (1.f + __expf(-z)));
    v[k] = yg; ss += yg * yg;
  }
  __shared__ float red[256];
  red[t] = ss;
  __syncthreads();
  for (int off = 128; off > 0; off >>= 1) {
    if (t < off) red[t] += red[t + off];
    __syncthreads();
  }
  const float r = rsqrtf(red[0] / (float)D_SSM + 1e-5f);
#pragma unroll
  for (int k = 0; k < 8; ++k) {
    const int ch = t + k * 256;
    yg16[g * D_SSM + ch] = (_Float16)(v[k] * r * norm_w[ch]);
  }
}

// =====================================================================
// host launcher
// =====================================================================
extern "C" void kernel_launch(void* const* d_in, const int* in_sizes, int n_in,
                              void* d_out, int out_size, void* d_ws, size_t ws_size,
                              hipStream_t stream)
{
  const float* u          = (const float*)d_in[0];
  const float* in_proj_w  = (const float*)d_in[1];
  const float* conv_w     = (const float*)d_in[2];
  const float* conv_b     = (const float*)d_in[3];
  const float* dt_bias    = (const float*)d_in[4];
  const float* A_log      = (const float*)d_in[5];
  const float* Dp         = (const float*)d_in[6];
  const float* norm_w     = (const float*)d_in[7];
  const float* out_proj_w = (const float*)d_in[8];
  float* out = (float*)d_out;

  char* ws = (char*)d_ws;
  size_t o = 0;
  float*    zxbcdt = (float*)(ws + o);    o += (size_t)BL * D_IN_PROJ * 4;        // 71.8 MB
  float*    dtws   = (float*)(ws + o);    o += (size_t)BL * NHEADS * 4;
  float*    dacs   = (float*)(ws + o);    o += (size_t)BATCH * NHEADS * NC * CHUNK * 4;
  float*    csum   = (float*)(ws + o);    o += (size_t)BATCH * NHEADS * NC * 4;
  float*    localS = (float*)(ws + o);    o += (size_t)BATCH * NC * NHEADS * 64 * 128 * 4;
  float*    yws    = (float*)(ws + o);    o += (size_t)BL * D_SSM * 4;            // 33.6 MB
  _Float16* u16    = (_Float16*)(ws + o); o += (size_t)BL * D_MODEL * 2;
  _Float16* wi16   = (_Float16*)(ws + o); o += (size_t)NPROJ_PAD * D_MODEL * 2;   // zero-padded
  _Float16* wo16   = (_Float16*)(ws + o); o += (size_t)D_MODEL * D_SSM * 2;
  _Float16* x16    = (_Float16*)(ws + o); o += (size_t)BL * D_SSM * 2;
  _Float16* bc16   = (_Float16*)(ws + o); o += (size_t)BL * 256 * 2;
  _Float16* xdt16  = (_Float16*)(ws + o); o += (size_t)BATCH * NC * NHEADS * 64 * 256 * 2;
  _Float16* prev16 = (_Float16*)(ws + o); o += (size_t)BATCH * NC * NHEADS * 64 * 128 * 2;
  _Float16* yg16   = (_Float16*)(ws + o); o += (size_t)BL * D_SSM * 2;

  dim3 blk(256);
  // 0) f16 operand prep (in_proj padded with zero rows -> guard-free GEMM)
  {
    const long nu = (long)BL * D_MODEL;
    k_cvt<<<(nu + 255) / 256, blk, 0, stream>>>(u, u16, nu, nu);
    const long nwi = (long)D_IN_PROJ * D_MODEL, nwip = (long)NPROJ_PAD * D_MODEL;
    k_cvt<<<(nwip + 255) / 256, blk, 0, stream>>>(in_proj_w, wi16, nwi, nwip);
    const long nwo = (long)D_MODEL * D_SSM;
    k_cvt<<<(nwo + 255) / 256, blk, 0, stream>>>(out_proj_w, wo16, nwo, nwo);
  }
  // 1) in_proj GEMM (TDM-staged)
  k_gemm16<<<dim3(NPROJ_PAD / 128, BL / 128), blk, 0, stream>>>(
      u16, wi16, zxbcdt, BL, D_IN_PROJ, D_MODEL);
  // 2) depthwise causal conv + SiLU -> f16 x / B|C
  k_conv<<<(BL * CONV_DIM) / 256, blk, 0, stream>>>(zxbcdt, conv_w, conv_b, x16, bc16);
  // 3) dt softplus + per-chunk cumsum of dA
  k_dt_scan<<<BATCH * NHEADS * NC, blk, 0, stream>>>(zxbcdt, dt_bias, A_log,
                                                     dtws, dacs, csum);
  // 4) Xdt transposed f16 layout
  k_xdtT<<<(BATCH * NC * NHEADS * 64 * 256) / 256, blk, 0, stream>>>(x16, dtws, xdt16);
  // 5) local chunk states (WMMA)
  k_chunk_state<<<BATCH * NC * NHEADS, blk, 0, stream>>>(bc16, xdt16, dacs, csum, localS);
  // 6) inter-chunk scan -> f16 prev states
  k_state_scan<<<BATCH * NHEADS, blk, 0, stream>>>(localS, csum, prev16);
  // 7) intra-chunk output (3 WMMA passes, async-DMA staging)
  k_chunk_y<<<BATCH * NC * NHEADS * 4, blk, 0, stream>>>(bc16, xdt16, prev16, x16,
                                                         dacs, Dp, yws);
  // 8) gate + RMSNorm -> f16 activations
  k_gate_norm<<<BL, blk, 0, stream>>>(yws, zxbcdt, norm_w, yg16);
  // 9) out_proj GEMM (TDM-staged)
  k_gemm16<<<dim3(D_MODEL / 128, BL / 128), blk, 0, stream>>>(
      yg16, wo16, out, BL, D_MODEL, D_SSM);
}
